// Qwen3Attention_4054449128306
// MI455X (gfx1250) — matrix operand order repaired and checked by reading, not compiled, not run on hardware
//
#include <hip/hip_runtime.h>

// ---------------------------------------------------------------------------
// Qwen3 attention block for MI455X (gfx1250), bf16 WMMA pipeline.
//   hidden(f32) -> bf16 -> [Q/K/V GEMM, wmma bf16] -> f32
//   -> RMSnorm + RoPE (+1/sqrt(D) folded into Q) -> bf16 (B,H,S,D)
//   -> flash attention (wmma bf16, online softmax, TDM tile loads) -> bf16
//   -> O GEMM (wmma bf16) -> f32 d_out
// ---------------------------------------------------------------------------

typedef unsigned short u16;
typedef unsigned int   u32;

static constexpr int BATCH   = 2;
static constexpr int SEQ     = 2048;
static constexpr int HIDDEN  = 2048;
static constexpr int NHEADS  = 16;
static constexpr int NKV     = 8;
static constexpr int HDIM    = 128;
static constexpr int BS      = BATCH * SEQ;          // 4096 rows
static constexpr float NEGV  = -1.0e9f;

struct __align__(16) U4 { u32 x, y, z, w; };

typedef __attribute__((ext_vector_type(16))) __bf16 v16bf;
typedef __attribute__((ext_vector_type(8)))  __bf16 v8bf;
typedef __attribute__((ext_vector_type(8)))  float  v8f;

union BF8 { U4 u; v8bf v; };

__device__ __forceinline__ v8bf ld8(const u16* p) {
    BF8 t; t.u = *reinterpret_cast<const U4*>(p); return t.v;
}
__device__ __forceinline__ v16bf cat16(v8bf lo, v8bf hi) {
    return __builtin_shufflevector(lo, hi, 0,1,2,3,4,5,6,7,8,9,10,11,12,13,14,15);
}
__device__ __forceinline__ u16 f2bf(float f) {         // round-to-nearest-even
    u32 u = __float_as_uint(f);
    return (u16)((u + 0x7FFFu + ((u >> 16) & 1u)) >> 16);
}

// ---------------------------------------------------------------------------
// Tensor Data Mover (gfx1250). This toolchain exposes the 6-arg builtin:
//   (uint32x4 g0, int32x8 g1, int32x4 g2, int32x4 g3, int32x8, i32 cpol)
// Guarded so absence falls back to manual copies.
// ---------------------------------------------------------------------------
#if __has_builtin(__builtin_amdgcn_tensor_load_to_lds) && \
    __has_builtin(__builtin_amdgcn_s_wait_tensorcnt)
#define USE_TDM 1
#else
#define USE_TDM 0
#endif

#if USE_TDM
typedef __attribute__((ext_vector_type(4))) unsigned int u32x4;
typedef __attribute__((ext_vector_type(8))) int          i32x8;
typedef __attribute__((ext_vector_type(4))) int          i32x4;

// 2D tile load, 2-byte elements, with LDS row padding.
//   tensor_d0/d1  : tensor extents (elements) for OOB
//   stride_d0     : elements between consecutive rows in memory
//   tile_d0/d1    : tile extents (d0 = contiguous)
//   pad_iv, pad_am: D# pad_interval / pad_amount codes (see ISA 08 §8.4)
__device__ __forceinline__ void tdm_load_2d(
    const u16* gaddr, u32 lds_off,
    u32 tensor_d0, u32 tensor_d1, u32 stride_d0,
    u32 tile_d0, u32 tile_d1, u32 pad_iv, u32 pad_am)
{
    unsigned long long ga = (unsigned long long)(size_t)gaddr;
    u32x4 g0;
    g0[0] = 1u;                                        // count=1, user mode
    g0[1] = lds_off;                                   // lds_addr
    g0[2] = (u32)(ga & 0xFFFFFFFFu);                   // global_addr[31:0]
    g0[3] = (u32)((ga >> 32) & 0x01FFFFFFu) | (2u << 30); // [56:32] | type=2
    i32x8 g1;
    g1[0] = (int)((1u << 16) | (1u << 20) |            // data_size=2B, pad_en
                  (pad_iv << 22) | (pad_am << 25));
    g1[1] = (int)((tensor_d0 & 0xFFFFu) << 16);        // td0[15:0] @ bits63:48
    g1[2] = (int)(((tensor_d0 >> 16) & 0xFFFFu) |      // td0[31:16]
                  ((tensor_d1 & 0xFFFFu) << 16));      // td1[15:0]
    g1[3] = (int)(((tensor_d1 >> 16) & 0xFFFFu) |      // td1[31:16]
                  ((tile_d0 & 0xFFFFu) << 16));        // tile_dim0
    g1[4] = (int)(tile_d1 & 0xFFFFu);                  // tile_dim1, tile_dim2=0
    g1[5] = (int)stride_d0;                            // dim0_stride[31:0]
    g1[6] = 0;                                         // stride hi / dim1_stride lo
    g1[7] = 0;
    i32x4 z4; z4[0] = 0; z4[1] = 0; z4[2] = 0; z4[3] = 0;
    i32x8 z8;
    #pragma unroll
    for (int i = 0; i < 8; ++i) z8[i] = 0;
    __builtin_amdgcn_tensor_load_to_lds(g0, g1, z4, z4, z8, 0);
}
#endif

// ---------------------------------------------------------------------------
// Generic f32 -> bf16 convert
// ---------------------------------------------------------------------------
__global__ void f32_to_bf16(const float* __restrict__ in, u16* __restrict__ out, int n) {
    int i = blockIdx.x * blockDim.x + threadIdx.x;
    if (i < n) out[i] = f2bf(in[i]);
}

// V: (B,S,KV,D) f32 -> (B,KV,D,S) bf16 (transposed so P*V B-frags are contiguous)
__global__ void v_transpose_bf16(const float* __restrict__ Vf, u16* __restrict__ Vt) {
    int i = blockIdx.x * blockDim.x + threadIdx.x;
    if (i >= BATCH * SEQ * NKV * HDIM) return;
    int d  =  i        & 127;
    int kv = (i >> 7)  & 7;
    int s  = (i >> 10) & 2047;
    int b  =  i >> 21;
    float v = Vf[(size_t)(b * SEQ + s) * (NKV * HDIM) + kv * HDIM + d];
    Vt[((size_t)(b * NKV + kv) * HDIM + d) * SEQ + s] = f2bf(v);
}

// ---------------------------------------------------------------------------
// GEMM: C[M,N] = A[M,K] * W[N,K]^T   (bf16 in, f32 out)
// 256 threads = 8 waves; tile 128x128x32; wave w -> rows w*16..w*16+15.
// Software-pipelined: next K-tile global loads issued before WMMAs of the
// current tile; B-frags preloaded in groups of 4 so WMMAs are not fenced
// one-by-one by s_wait_dscnt.
// ---------------------------------------------------------------------------
__global__ __launch_bounds__(256) void gemm_bf16_nt(
    const u16* __restrict__ A, const u16* __restrict__ W,
    float* __restrict__ C, int M, int N, int K)
{
    __shared__ u16 As[128][40];   // +8 pad, 16B-aligned rows
    __shared__ u16 Ws[128][40];

    const int tid  = threadIdx.x;
    const int lane = tid & 31, wv = tid >> 5;
    const int half = lane >> 4, mrow = lane & 15;
    const int n0 = blockIdx.x * 128;
    const int m0 = blockIdx.y * 128;

    // this thread's two 16B copy slots (i = tid, tid+256)
    const int r0 = tid >> 2,          c0 = (tid & 3) << 3;
    const int r1 = (tid + 256) >> 2,  c1 = ((tid + 256) & 3) << 3;
    const u16* Ap0 = A + (size_t)(m0 + r0) * K + c0;
    const u16* Ap1 = A + (size_t)(m0 + r1) * K + c1;
    const u16* Wp0 = W + (size_t)(n0 + r0) * K + c0;
    const u16* Wp1 = W + (size_t)(n0 + r1) * K + c1;

    v8f acc[8];
    #pragma unroll
    for (int j = 0; j < 8; ++j)
        #pragma unroll
        for (int e = 0; e < 8; ++e) acc[j][e] = 0.0f;

    U4 ra0 = *reinterpret_cast<const U4*>(Ap0);
    U4 ra1 = *reinterpret_cast<const U4*>(Ap1);
    U4 rw0 = *reinterpret_cast<const U4*>(Wp0);
    U4 rw1 = *reinterpret_cast<const U4*>(Wp1);

    for (int kk = 0; kk < K; kk += 32) {
        __syncthreads();                        // previous tile fully consumed
        *reinterpret_cast<U4*>(&As[r0][c0]) = ra0;
        *reinterpret_cast<U4*>(&As[r1][c1]) = ra1;
        *reinterpret_cast<U4*>(&Ws[r0][c0]) = rw0;
        *reinterpret_cast<U4*>(&Ws[r1][c1]) = rw1;
        __syncthreads();

        if (kk + 32 < K) {                      // prefetch next tile (overlaps WMMA)
            ra0 = *reinterpret_cast<const U4*>(Ap0 + kk + 32);
            ra1 = *reinterpret_cast<const U4*>(Ap1 + kk + 32);
            rw0 = *reinterpret_cast<const U4*>(Wp0 + kk + 32);
            rw1 = *reinterpret_cast<const U4*>(Wp1 + kk + 32);
        }

        // A-frag: lane m = lane%16, K 8h..8h+7 and 16+8h..16+8h+7 per half-group h
        v16bf a = cat16(ld8(&As[wv * 16 + mrow][half * 8]),
                        ld8(&As[wv * 16 + mrow][16 + half * 8]));
        v16bf bfr[4];
        #pragma unroll
        for (int j = 0; j < 4; ++j)
            bfr[j] = cat16(ld8(&Ws[j * 16 + mrow][half * 8]),
                           ld8(&Ws[j * 16 + mrow][16 + half * 8]));
        #pragma unroll
        for (int j = 0; j < 4; ++j)
            acc[j] = __builtin_amdgcn_wmma_f32_16x16x32_bf16(
                false, a, false, bfr[j], (short)0, acc[j], false, false);
        #pragma unroll
        for (int j = 0; j < 4; ++j)
            bfr[j] = cat16(ld8(&Ws[(j + 4) * 16 + mrow][half * 8]),
                           ld8(&Ws[(j + 4) * 16 + mrow][16 + half * 8]));
        #pragma unroll
        for (int j = 0; j < 4; ++j)
            acc[j + 4] = __builtin_amdgcn_wmma_f32_16x16x32_bf16(
                false, a, false, bfr[j], (short)0, acc[j + 4], false, false);
    }

    #pragma unroll
    for (int j = 0; j < 8; ++j)
        #pragma unroll
        for (int r = 0; r < 8; ++r) {
            int gm = m0 + wv * 16 + r + 8 * half;   // C-layout: VGPR r -> M=r / r+8
            int gn = n0 + j * 16 + mrow;
            C[(size_t)gm * N + gn] = acc[j][r];
        }
}

// ---------------------------------------------------------------------------
// Fused per-head RMS norm + RoPE.
// grid: (B*S, NHEADS+NKV), block 128 (one thread per dim).
// ---------------------------------------------------------------------------
__global__ __launch_bounds__(128) void rmsrope_kernel(
    const float* __restrict__ Qf, const float* __restrict__ Kf,
    const float* __restrict__ qw, const float* __restrict__ kw,
    const int*   __restrict__ pos_ids,
    u16* __restrict__ Qo, u16* __restrict__ Ko)
{
    __shared__ float red[4];
    __shared__ float xs[128];

    const int bs = blockIdx.x;
    const int b  = bs >> 11, s = bs & 2047;
    const int y  = blockIdx.y;
    const int d  = threadIdx.x;
    const bool isq = (y < NHEADS);
    const int head = isq ? y : y - NHEADS;

    float x, w;
    if (isq) { x = Qf[(size_t)bs * HIDDEN + head * HDIM + d];       w = qw[d]; }
    else     { x = Kf[(size_t)bs * (NKV * HDIM) + head * HDIM + d]; w = kw[d]; }

    float ss = x * x;
    #pragma unroll
    for (int off = 1; off < 32; off <<= 1) ss += __shfl_xor(ss, off, 32);
    if ((threadIdx.x & 31) == 0) red[threadIdx.x >> 5] = ss;
    __syncthreads();
    float tot = red[0] + red[1] + red[2] + red[3];
    float rms = rsqrtf(tot * (1.0f / 128.0f) + 1e-6f);
    float xn  = x * rms * w;
    xs[d] = xn;
    __syncthreads();
    float part = xs[d ^ 64];

    float pos = (float)pos_ids[bs];
    int   d2  = d & 63;
    float invf = __expf((float)d2 * (-13.815510558f / 64.0f)); // 1e6^(-2d/128)
    float th = pos * invf;
    float c = __cosf(th), sn = __sinf(th);
    float out = (d < 64) ? (xn * c - part * sn) : (xn * c + part * sn);
    if (isq) {
        out *= 0.08838834764831845f;   // 1/sqrt(128) folded into Q
        Qo[((size_t)(b * NHEADS + head) * SEQ + s) * HDIM + d] = f2bf(out);
    } else {
        Ko[((size_t)(b * NKV + head) * SEQ + s) * HDIM + d] = f2bf(out);
    }
}

// ---------------------------------------------------------------------------
// Flash attention, causal. block = 256 threads (8 waves), each wave 16 q-rows,
// block covers 128 q-rows of one (b, h); 8 waves share each K/V tile.
// Key blocks of 32 keys; K/V tiles staged by the TDM when available.
// grid: (S/128, NHEADS, B).
// ---------------------------------------------------------------------------
__global__ __launch_bounds__(256) void attn_kernel(
    const u16* __restrict__ Q,    // (B,H,S,D)  bf16, pre-scaled
    const u16* __restrict__ Kt,   // (B,KV,S,D) bf16
    const u16* __restrict__ Vt,   // (B,KV,D,S) bf16 (transposed)
    u16* __restrict__ O)          // (B,S,H*D)  bf16
{
    __shared__ u16 kt[32][136];      // 32 keys x 128 dims (+8 pad) = 64+4 DW rows
    __shared__ u16 vtT[128][40];     // 128 dims x 32 keys (+8 pad) = 16+4 DW rows
    __shared__ u16 pt[8][16][40];    // per-wave P tile: 16 rows x 32 keys

    const int tid  = threadIdx.x;
    const int lane = tid & 31, wv = tid >> 5;
    const int half = lane >> 4, mrow = lane & 15;
    const int qblk = blockIdx.x, h = blockIdx.y, b = blockIdx.z;
    const int kv = h >> 1;                       // G = NHEADS/NKV = 2
    const int q0 = qblk * 128 + wv * 16;

    const u16* kbase = Kt + (size_t)(b * NKV + kv) * SEQ * HDIM;
    const u16* vbase = Vt + (size_t)(b * NKV + kv) * HDIM * SEQ;

    // Q fragments for 4 K-chunks of 32 dims, register-resident for whole loop
    const u16* qrow = Q + ((size_t)(b * NHEADS + h) * SEQ + q0 + mrow) * HDIM;
    v16bf qf[4];
    #pragma unroll
    for (int c = 0; c < 4; ++c)
        qf[c] = cat16(ld8(qrow + c * 32 + half * 8),
                      ld8(qrow + c * 32 + 16 + half * 8));

    v8f acc[8];
    float mi[8], li[8];
    #pragma unroll
    for (int r = 0; r < 8; ++r) {
        mi[r] = -3.0e38f; li[r] = 0.0f;
        #pragma unroll
        for (int j = 0; j < 8; ++j) acc[j][r] = 0.0f;
    }

    const int nkb = qblk * 4 + 4;                // causal: keys <= q0_block+127
    for (int kb = 0; kb < nkb; ++kb) {
        __syncthreads();                         // prior tile fully consumed
#if USE_TDM
        if (wv == 0) {
            // K tile: 32 rows x 128 elems, row stride 128; LDS pad 4 DW / 64 DW
            tdm_load_2d(kbase + (size_t)(kb * 32) * HDIM,
                        (u32)(size_t)&kt[0][0],
                        /*td0=*/HDIM, /*td1=*/32, /*stride=*/HDIM,
                        /*tile_d0=*/HDIM, /*tile_d1=*/32,
                        /*pad_iv=*/5, /*pad_am=*/3);
            // V tile: 128 rows x 32 elems, row stride SEQ; LDS pad 4 DW / 16 DW
            tdm_load_2d(vbase + kb * 32,
                        (u32)(size_t)&vtT[0][0],
                        /*td0=*/SEQ, /*td1=*/HDIM, /*stride=*/SEQ,
                        /*tile_d0=*/32, /*tile_d1=*/HDIM,
                        /*pad_iv=*/3, /*pad_am=*/3);
            __builtin_amdgcn_s_wait_tensorcnt(0);
        }
#else
        for (int i = tid; i < 512; i += 256) {
            int key = i >> 4, c8 = (i & 15) << 3;
            *reinterpret_cast<U4*>(&kt[key][c8]) = *reinterpret_cast<const U4*>(
                kbase + ((size_t)(kb * 32 + key)) * HDIM + c8);
        }
        for (int i = tid; i < 512; i += 256) {
            int d = i >> 2, k8 = (i & 3) << 3;
            *reinterpret_cast<U4*>(&vtT[d][k8]) = *reinterpret_cast<const U4*>(
                vbase + (size_t)d * SEQ + kb * 32 + k8);
        }
#endif
        __syncthreads();

        // ---- scores + online softmax, two 16-key subtiles ----
        #pragma unroll
        for (int sub = 0; sub < 2; ++sub) {
            // two independent accumulation chains to break the WMMA RAW chain
            v8f s0, s1;
            #pragma unroll
            for (int e = 0; e < 8; ++e) { s0[e] = 0.0f; s1[e] = 0.0f; }
            v16bf kf0 = cat16(ld8(&kt[sub * 16 + mrow][0 * 32 + half * 8]),
                              ld8(&kt[sub * 16 + mrow][0 * 32 + 16 + half * 8]));
            v16bf kf1 = cat16(ld8(&kt[sub * 16 + mrow][1 * 32 + half * 8]),
                              ld8(&kt[sub * 16 + mrow][1 * 32 + 16 + half * 8]));
            v16bf kf2 = cat16(ld8(&kt[sub * 16 + mrow][2 * 32 + half * 8]),
                              ld8(&kt[sub * 16 + mrow][2 * 32 + 16 + half * 8]));
            v16bf kf3 = cat16(ld8(&kt[sub * 16 + mrow][3 * 32 + half * 8]),
                              ld8(&kt[sub * 16 + mrow][3 * 32 + 16 + half * 8]));
            s0 = __builtin_amdgcn_wmma_f32_16x16x32_bf16(false, qf[0], false, kf0, (short)0, s0, false, false);
            s1 = __builtin_amdgcn_wmma_f32_16x16x32_bf16(false, qf[1], false, kf1, (short)0, s1, false, false);
            s0 = __builtin_amdgcn_wmma_f32_16x16x32_bf16(false, qf[2], false, kf2, (short)0, s0, false, false);
            s1 = __builtin_amdgcn_wmma_f32_16x16x32_bf16(false, qf[3], false, kf3, (short)0, s1, false, false);

            const int kj = kb * 32 + sub * 16 + mrow;
            #pragma unroll
            for (int r = 0; r < 8; ++r) {
                int qi = q0 + r + 8 * half;
                float val = s0[r] + s1[r];
                if (kj > qi) val = NEGV;
                float rmax = val;
                #pragma unroll
                for (int off = 1; off < 16; off <<= 1)
                    rmax = fmaxf(rmax, __shfl_xor(rmax, off, 32));
                float mold = mi[r];
                float mnew = fmaxf(mold, rmax);
                float p    = __expf(val - mnew);
                float rsum = p;
                #pragma unroll
                for (int off = 1; off < 16; off <<= 1)
                    rsum += __shfl_xor(rsum, off, 32);
                float alpha = __expf(mold - mnew);
                li[r] = li[r] * alpha + rsum;
                mi[r] = mnew;
                #pragma unroll
                for (int j = 0; j < 8; ++j) acc[j][r] *= alpha;
                pt[wv][r + 8 * half][sub * 16 + mrow] = f2bf(p);   // C-layout -> LDS
            }
        }

        // ---- P (16x32) * V (32x128), B-frags preloaded in groups of 4 ----
        v16bf pf = cat16(ld8(&pt[wv][mrow][half * 8]),
                         ld8(&pt[wv][mrow][16 + half * 8]));
        v16bf vfr[4];
        #pragma unroll
        for (int j = 0; j < 4; ++j)
            vfr[j] = cat16(ld8(&vtT[j * 16 + mrow][half * 8]),
                           ld8(&vtT[j * 16 + mrow][16 + half * 8]));
        #pragma unroll
        for (int j = 0; j < 4; ++j)
            acc[j] = __builtin_amdgcn_wmma_f32_16x16x32_bf16(
                false, pf, false, vfr[j], (short)0, acc[j], false, false);
        #pragma unroll
        for (int j = 0; j < 4; ++j)
            vfr[j] = cat16(ld8(&vtT[(j + 4) * 16 + mrow][half * 8]),
                           ld8(&vtT[(j + 4) * 16 + mrow][16 + half * 8]));
        #pragma unroll
        for (int j = 0; j < 4; ++j)
            acc[j + 4] = __builtin_amdgcn_wmma_f32_16x16x32_bf16(
                false, pf, false, vfr[j], (short)0, acc[j + 4], false, false);
    }

    // normalize + write bf16 (B,S,H*D)
    #pragma unroll
    for (int j = 0; j < 8; ++j)
        #pragma unroll
        for (int r = 0; r < 8; ++r) {
            int sidx = q0 + r + 8 * half;
            int col  = h * HDIM + j * 16 + mrow;
            O[((size_t)b * SEQ + sidx) * (NHEADS * HDIM) + col] =
                f2bf(acc[j][r] / li[r]);
        }
}

// ---------------------------------------------------------------------------
// Launch
// ---------------------------------------------------------------------------
extern "C" void kernel_launch(void* const* d_in, const int* in_sizes, int n_in,
                              void* d_out, int out_size, void* d_ws, size_t ws_size,
                              hipStream_t stream) {
    const float* hs  = (const float*)d_in[0];
    // d_in[1] attention_mask: causal, recomputed in-kernel
    const int*   pos = (const int*)d_in[2];
    const float* q_w = (const float*)d_in[3];
    const float* k_w = (const float*)d_in[4];
    const float* v_w = (const float*)d_in[5];
    const float* o_w = (const float*)d_in[6];
    const float* qnw = (const float*)d_in[7];
    const float* knw = (const float*)d_in[8];

    char* ws = (char*)d_ws;
    size_t off = 0;
    auto alloc = [&](size_t bytes) -> void* {
        void* p = ws + off;
        off = (off + bytes + 255) & ~(size_t)255;
        return p;
    };

    u16*   hs_bf = (u16*)alloc((size_t)BS * HIDDEN * 2);
    u16*   qw_bf = (u16*)alloc((size_t)HIDDEN * HIDDEN * 2);
    u16*   kw_bf = (u16*)alloc((size_t)NKV * HDIM * HIDDEN * 2);
    u16*   vw_bf = (u16*)alloc((size_t)NKV * HDIM * HIDDEN * 2);
    u16*   ow_bf = (u16*)alloc((size_t)HIDDEN * HIDDEN * 2);
    float* qf    = (float*)alloc((size_t)BS * HIDDEN * 4);
    float* kf    = (float*)alloc((size_t)BS * NKV * HDIM * 4);
    float* vf    = (float*)alloc((size_t)BS * NKV * HDIM * 4);
    u16*   qbf   = (u16*)alloc((size_t)BS * HIDDEN * 2);       // (B,H,S,D)
    u16*   kbf   = (u16*)alloc((size_t)BS * NKV * HDIM * 2);   // (B,KV,S,D)
    u16*   vtb   = (u16*)alloc((size_t)BS * NKV * HDIM * 2);   // (B,KV,D,S)
    u16*   attn_bf = (u16*)qf;   // alias: qf is dead after rmsrope

    const int CT = 256;
    f32_to_bf16<<<(BS * HIDDEN + CT - 1) / CT, CT, 0, stream>>>(hs, hs_bf, BS * HIDDEN);
    f32_to_bf16<<<(HIDDEN * HIDDEN + CT - 1) / CT, CT, 0, stream>>>(q_w, qw_bf, HIDDEN * HIDDEN);
    f32_to_bf16<<<(NKV * HDIM * HIDDEN + CT - 1) / CT, CT, 0, stream>>>(k_w, kw_bf, NKV * HDIM * HIDDEN);
    f32_to_bf16<<<(NKV * HDIM * HIDDEN + CT - 1) / CT, CT, 0, stream>>>(v_w, vw_bf, NKV * HDIM * HIDDEN);
    f32_to_bf16<<<(HIDDEN * HIDDEN + CT - 1) / CT, CT, 0, stream>>>(o_w, ow_bf, HIDDEN * HIDDEN);

    // projections: C = X * W^T
    gemm_bf16_nt<<<dim3(HIDDEN / 128, BS / 128), 256, 0, stream>>>(hs_bf, qw_bf, qf, BS, HIDDEN, HIDDEN);
    gemm_bf16_nt<<<dim3(NKV * HDIM / 128, BS / 128), 256, 0, stream>>>(hs_bf, kw_bf, kf, BS, NKV * HDIM, HIDDEN);
    gemm_bf16_nt<<<dim3(NKV * HDIM / 128, BS / 128), 256, 0, stream>>>(hs_bf, vw_bf, vf, BS, NKV * HDIM, HIDDEN);

    rmsrope_kernel<<<dim3(BS, NHEADS + NKV), 128, 0, stream>>>(qf, kf, qnw, knw, pos, qbf, kbf);
    v_transpose_bf16<<<(BATCH * SEQ * NKV * HDIM + CT - 1) / CT, CT, 0, stream>>>(vf, vtb);

    attn_kernel<<<dim3(SEQ / 128, NHEADS, BATCH), 256, 0, stream>>>(qbf, kbf, vtb, attn_bf);

    // output projection straight into d_out (f32)
    gemm_bf16_nt<<<dim3(HIDDEN / 128, BS / 128), 256, 0, stream>>>(attn_bf, ow_bf, (float*)d_out,
                                                                   BS, HIDDEN, HIDDEN);
    (void)in_sizes; (void)n_in; (void)out_size; (void)ws_size;
}
